// EnBaseLayer_23880018166020
// MI455X (gfx1250) — matrix operand order repaired and checked by reading, not compiled, not run on hardware
//
#include <hip/hip_runtime.h>
#include <stdint.h>

#define NN 50000
#define NE 640000
#define HIDN 128
#define EF 4
#define NG 20
#define KZ 288          // 2*128 + 20 + 4 = 280, padded to multiple of 32
#define ETILE 64
#define NTILE 64

typedef __attribute__((ext_vector_type(16))) __bf16 v16bf;
typedef __attribute__((ext_vector_type(8)))  float  v8f;

union Frag { uint4 u[2]; v16bf v; };

__device__ __forceinline__ v8f wmma_bf16(const Frag& a, const Frag& b, v8f c) {
    return __builtin_amdgcn_wmma_f32_16x16x32_bf16(false, a.v, false, b.v,
                                                   (short)0, c, false, false);
}

__device__ __forceinline__ float silu_f(float v) { return v / (1.f + __expf(-v)); }

// ---------------------------------------------------------------------------
// Zero fill (workspace accumulators must be zeroed every launch)
// ---------------------------------------------------------------------------
__global__ void zero_kernel(float* p, int n) {
    int i = blockIdx.x * blockDim.x + threadIdx.x;
    if (i < n) p[i] = 0.f;
}

// ---------------------------------------------------------------------------
// Pack a row-major fp32 weight [K x N] into bf16 WMMA B-fragments.
// Tile (kt, nt) covers K in [kt*32, kt*32+32), N in [nt*16, nt*16+16).
// Storage: out[((kt*NT + nt)*32 + lane)*8 + v] = {W[k0][n], W[k0+1][n]} bf16x2
// where n = nt*16 + (lane&15), k0 = kt*32 + ((v>=4)?16:0) + ((lane>=16)?8:0) + 2*(v&3).
// ---------------------------------------------------------------------------
__global__ void pack_w(const float* __restrict__ W, uint32_t* __restrict__ out,
                       int K, int N, int KT, int NT) {
    int idx = blockIdx.x * blockDim.x + threadIdx.x;
    int total = KT * NT * 32 * 8;
    if (idx >= total) return;
    int v    = idx & 7;
    int lane = (idx >> 3) & 31;
    int tile = idx >> 8;
    int nt = tile % NT;
    int kt = tile / NT;
    int n  = nt * 16 + (lane & 15);
    int k0 = kt * 32 + ((v >> 2) << 4) + ((lane >> 4) << 3) + 2 * (v & 3);
    float a = (k0     < K) ? W[(size_t)k0       * N + n] : 0.f;
    float b = (k0 + 1 < K) ? W[(size_t)(k0 + 1) * N + n] : 0.f;
    union { __bf16 h[2]; uint32_t u; } p;
    p.h[0] = (__bf16)a;
    p.h[1] = (__bf16)b;
    out[idx] = p.u;
}

// ---------------------------------------------------------------------------
// Edge kernel: 128 threads (4 waves), 64 edges per block.
// ---------------------------------------------------------------------------
__global__ __launch_bounds__(128) void edge_kernel(
    const float* __restrict__ h, const float* __restrict__ x,
    const int* __restrict__ ei, const float* __restrict__ eattr,
    const float* __restrict__ be1, const float* __restrict__ be2,
    const float* __restrict__ Winf, const float* __restrict__ binf,
    const float* __restrict__ bx1, const float* __restrict__ Wx2,
    const uint32_t* __restrict__ pWe1, const uint32_t* __restrict__ pWe2,
    const uint32_t* __restrict__ pWx1,
    float* __restrict__ mi, float* __restrict__ dxb)
{
    __shared__ __align__(16) unsigned char buf[49152];   // staged arena
    __shared__ float s_rel[ETILE][3];
    __shared__ float s_d[ETILE];
    __shared__ int   s_src[ETILE];
    __shared__ int   s_dst[ETILE];
    __shared__ float s_eij[ETILE];

    const int t    = threadIdx.x;
    const int lane = t & 31;
    const int w    = t >> 5;
    const int g    = lane >> 4;
    const int m    = lane & 15;
    const int rowA = w * 16 + m;
    const int e0   = blockIdx.x * ETILE;

    __bf16* zbf = (__bf16*)buf;                 // [ETILE][KZ]

    // ---- phase 0: indices, rel_x, distance, gaussian smear, edge_attr ----
    if (t < ETILE) {
        const int e   = e0 + t;
        const int src = ei[e];
        const int dst = ei[NE + e];
        s_src[t] = src;
        s_dst[t] = dst;
        float rx = x[dst * 3 + 0] - x[src * 3 + 0];
        float ry = x[dst * 3 + 1] - x[src * 3 + 1];
        float rz = x[dst * 3 + 2] - x[src * 3 + 2];
        s_rel[t][0] = rx; s_rel[t][1] = ry; s_rel[t][2] = rz;
        float d = sqrtf(rx * rx + ry * ry + rz * rz + 1e-8f);
        s_d[t] = d;
        __bf16* row = zbf + t * KZ;
        const float step  = 10.f / 19.f;
        const float coeff = -0.5f / (step * step);
        #pragma unroll
        for (int i2 = 0; i2 < NG; i2++) {
            float del = d - step * (float)i2;
            row[2 * HIDN + i2] = (__bf16)__expf(coeff * del * del);
        }
        #pragma unroll
        for (int c = 0; c < EF; c++)
            row[2 * HIDN + NG + c] = (__bf16)eattr[(size_t)e * EF + c];
        #pragma unroll
        for (int k = 2 * HIDN + NG + EF; k < KZ; k++) row[k] = (__bf16)0.f;
    }
    __syncthreads();

    // ---- phase 1: gather h[dst], h[src] -> bf16 z rows ----
    {
        const int e  = t >> 1;
        const int c0 = (t & 1) * 64;
        const float* hi = h + (size_t)s_dst[e] * HIDN;
        const float* hj = h + (size_t)s_src[e] * HIDN;
        __bf16* row = zbf + e * KZ;
        for (int c = c0; c < c0 + 64; c++) {
            row[c]        = (__bf16)hi[c];
            row[HIDN + c] = (__bf16)hj[c];
        }
    }
    __syncthreads();

    const v8f zf = {0.f, 0.f, 0.f, 0.f, 0.f, 0.f, 0.f, 0.f};
    v8f acc[8];

    // ---- GEMM1: z[64x288] @ We1[288x128] ----
    #pragma unroll
    for (int nt = 0; nt < 8; nt++) acc[nt] = zf;
    for (int kt = 0; kt < 9; kt++) {
        Frag A;
        const __bf16* ap = zbf + rowA * KZ + kt * 32 + g * 8;
        A.u[0] = *(const uint4*)ap;
        A.u[1] = *(const uint4*)(ap + 16);
        const uint32_t* bp = pWe1 + (size_t)(kt * 8) * 256 + lane * 8;
        #pragma unroll
        for (int nt = 0; nt < 8; nt++) {
            Frag B;
            B.u[0] = *(const uint4*)bp;
            B.u[1] = *(const uint4*)(bp + 4);
            acc[nt] = wmma_bf16(A, B, acc[nt]);
            bp += 256;
        }
    }
    __syncthreads();
    __bf16* t1bf = (__bf16*)buf;                // [ETILE][HIDN] (reuses zbf)
    #pragma unroll
    for (int nt = 0; nt < 8; nt++) {
        const int n = nt * 16 + m;
        const float b1 = be1[n];
        #pragma unroll
        for (int v = 0; v < 8; v++) {
            const int r = w * 16 + v + 8 * g;
            t1bf[r * HIDN + n] = (__bf16)silu_f(acc[nt][v] + b1);
        }
    }
    __syncthreads();

    // ---- GEMM2: t1[64x128] @ We2[128x128] -> mij ----
    #pragma unroll
    for (int nt = 0; nt < 8; nt++) acc[nt] = zf;
    for (int kt = 0; kt < 4; kt++) {
        Frag A;
        const __bf16* ap = t1bf + rowA * HIDN + kt * 32 + g * 8;
        A.u[0] = *(const uint4*)ap;
        A.u[1] = *(const uint4*)(ap + 16);
        const uint32_t* bp = pWe2 + (size_t)(kt * 8) * 256 + lane * 8;
        #pragma unroll
        for (int nt = 0; nt < 8; nt++) {
            Frag B;
            B.u[0] = *(const uint4*)bp;
            B.u[1] = *(const uint4*)(bp + 4);
            acc[nt] = wmma_bf16(A, B, acc[nt]);
            bp += 256;
        }
    }
    __syncthreads();
    float*  mijf  = (float*)(buf + ETILE * HIDN * 2);   // [64][128] f32 at +16KB
    __bf16* mijbf = (__bf16*)buf;                       // [64][128] bf16 at +0
    #pragma unroll
    for (int nt = 0; nt < 8; nt++) {
        const int n = nt * 16 + m;
        const float b2 = be2[n];
        #pragma unroll
        for (int v = 0; v < 8; v++) {
            const int r = w * 16 + v + 8 * g;
            float val = silu_f(acc[nt][v] + b2);
            mijf[r * HIDN + n]  = val;
            mijbf[r * HIDN + n] = (__bf16)val;
        }
    }
    __syncthreads();

    // ---- eij = sigmoid(mij . Winf + binf) ----
    if (t < ETILE) {
        float s = binf[0];
        const float* row = mijf + t * HIDN;
        for (int n = 0; n < HIDN; n++) s += row[n] * Winf[n];
        s_eij[t] = 1.f / (1.f + __expf(-s));
    }
    __syncthreads();

    // ---- scatter: mi[dst] += mij * eij (L2-resident fp32 atomics) ----
    for (int i = t; i < ETILE * HIDN; i += 128) {
        const int e = i >> 7;
        const int n = i & (HIDN - 1);
        atomicAdd(&mi[(size_t)s_dst[e] * HIDN + n], mijf[i] * s_eij[e]);
    }
    __syncthreads();

    // ---- GEMM3: mij[64x128] @ Wx1[128x128] ----
    #pragma unroll
    for (int nt = 0; nt < 8; nt++) acc[nt] = zf;
    for (int kt = 0; kt < 4; kt++) {
        Frag A;
        const __bf16* ap = mijbf + rowA * HIDN + kt * 32 + g * 8;
        A.u[0] = *(const uint4*)ap;
        A.u[1] = *(const uint4*)(ap + 16);
        const uint32_t* bp = pWx1 + (size_t)(kt * 8) * 256 + lane * 8;
        #pragma unroll
        for (int nt = 0; nt < 8; nt++) {
            Frag B;
            B.u[0] = *(const uint4*)bp;
            B.u[1] = *(const uint4*)(bp + 4);
            acc[nt] = wmma_bf16(A, B, acc[nt]);
            bp += 256;
        }
    }
    float* t2f = (float*)(buf + ETILE * HIDN * 2);      // reuse mijf region
    #pragma unroll
    for (int nt = 0; nt < 8; nt++) {
        const int n = nt * 16 + m;
        const float b = bx1[n];
        #pragma unroll
        for (int v = 0; v < 8; v++) {
            const int r = w * 16 + v + 8 * g;
            t2f[r * HIDN + n] = silu_f(acc[nt][v] + b);
        }
    }
    __syncthreads();

    // ---- xg = tanh(t2 . Wx2); scatter rel_x/(d+1)*xg into dx[dst] ----
    if (t < ETILE) {
        const float* row = t2f + t * HIDN;
        float s = 0.f;
        for (int n = 0; n < HIDN; n++) s += row[n] * Wx2[n];
        const float xg = tanhf(s);
        const float c  = xg / (s_d[t] + 1.f);
        const int dst  = s_dst[t];
        atomicAdd(&dxb[dst * 3 + 0], s_rel[t][0] * c);
        atomicAdd(&dxb[dst * 3 + 1], s_rel[t][1] * c);
        atomicAdd(&dxb[dst * 3 + 2], s_rel[t][2] * c);
    }
}

// ---------------------------------------------------------------------------
// Node kernel: 128 threads (4 waves), 64 nodes per block.
// ---------------------------------------------------------------------------
__global__ __launch_bounds__(128) void node_kernel(
    const float* __restrict__ h, const float* __restrict__ x,
    const float* __restrict__ mask,
    const float* __restrict__ bn1, const float* __restrict__ bn2,
    const uint32_t* __restrict__ pWn1, const uint32_t* __restrict__ pWn2,
    const float* __restrict__ mi, const float* __restrict__ dxb,
    float* __restrict__ h_out, float* __restrict__ x_out)
{
    __shared__ __align__(16) unsigned char buf[NTILE * 256 * 2];   // 32KB arena

    const int t    = threadIdx.x;
    const int lane = t & 31;
    const int w    = t >> 5;
    const int g    = lane >> 4;
    const int m    = lane & 15;
    const int rowA = w * 16 + m;
    const int n0   = blockIdx.x * NTILE;

    __bf16* nzbf = (__bf16*)buf;                // [NTILE][256]

    // build nz = [mi, h] in bf16
    {
        const int r    = t >> 1;
        const int node = n0 + r;
        const bool valid = node < NN;
        const int c0 = (t & 1) * 64;
        __bf16* row = nzbf + r * 256;
        for (int c = c0; c < c0 + 64; c++) {
            row[c]       = valid ? (__bf16)mi[(size_t)node * HIDN + c] : (__bf16)0.f;
            row[128 + c] = valid ? (__bf16)h[(size_t)node * HIDN + c]  : (__bf16)0.f;
        }
    }
    __syncthreads();

    const v8f zf = {0.f, 0.f, 0.f, 0.f, 0.f, 0.f, 0.f, 0.f};
    v8f acc[8];

    // GEMM1: nz[64x256] @ Wn1[256x128]
    #pragma unroll
    for (int nt = 0; nt < 8; nt++) acc[nt] = zf;
    for (int kt = 0; kt < 8; kt++) {
        Frag A;
        const __bf16* ap = nzbf + rowA * 256 + kt * 32 + g * 8;
        A.u[0] = *(const uint4*)ap;
        A.u[1] = *(const uint4*)(ap + 16);
        const uint32_t* bp = pWn1 + (size_t)(kt * 8) * 256 + lane * 8;
        #pragma unroll
        for (int nt = 0; nt < 8; nt++) {
            Frag B;
            B.u[0] = *(const uint4*)bp;
            B.u[1] = *(const uint4*)(bp + 4);
            acc[nt] = wmma_bf16(A, B, acc[nt]);
            bp += 256;
        }
    }
    __syncthreads();
    __bf16* t1bf = (__bf16*)buf;                // [NTILE][128]
    #pragma unroll
    for (int nt = 0; nt < 8; nt++) {
        const int n = nt * 16 + m;
        const float b1 = bn1[n];
        #pragma unroll
        for (int v = 0; v < 8; v++) {
            const int r = w * 16 + v + 8 * g;
            t1bf[r * HIDN + n] = (__bf16)silu_f(acc[nt][v] + b1);
        }
    }
    __syncthreads();

    // GEMM2: t1[64x128] @ Wn2[128x128]
    #pragma unroll
    for (int nt = 0; nt < 8; nt++) acc[nt] = zf;
    for (int kt = 0; kt < 4; kt++) {
        Frag A;
        const __bf16* ap = t1bf + rowA * HIDN + kt * 32 + g * 8;
        A.u[0] = *(const uint4*)ap;
        A.u[1] = *(const uint4*)(ap + 16);
        const uint32_t* bp = pWn2 + (size_t)(kt * 8) * 256 + lane * 8;
        #pragma unroll
        for (int nt = 0; nt < 8; nt++) {
            Frag B;
            B.u[0] = *(const uint4*)bp;
            B.u[1] = *(const uint4*)(bp + 4);
            acc[nt] = wmma_bf16(A, B, acc[nt]);
            bp += 256;
        }
    }

    // epilogue: h_out = h + (acc + bn2)
    #pragma unroll
    for (int nt = 0; nt < 8; nt++) {
        const int n = nt * 16 + m;
        const float b2 = bn2[n];
        #pragma unroll
        for (int v = 0; v < 8; v++) {
            const int r    = w * 16 + v + 8 * g;
            const int node = n0 + r;
            if (node < NN) {
                const size_t idx = (size_t)node * HIDN + n;
                h_out[idx] = h[idx] + acc[nt][v] + b2;
            }
        }
    }

    // x_out = x + dx * mask
    if (t < NTILE) {
        const int node = n0 + t;
        if (node < NN) {
            const float mk = mask[node];
            #pragma unroll
            for (int c = 0; c < 3; c++)
                x_out[node * 3 + c] = x[node * 3 + c] + dxb[node * 3 + c] * mk;
        }
    }
}

// ---------------------------------------------------------------------------
extern "C" void kernel_launch(void* const* d_in, const int* in_sizes, int n_in,
                              void* d_out, int out_size, void* d_ws, size_t ws_size,
                              hipStream_t stream)
{
    const float* h     = (const float*)d_in[0];
    const float* x     = (const float*)d_in[1];
    const int*   ei    = (const int*)d_in[2];
    const float* mask  = (const float*)d_in[3];
    const float* eattr = (const float*)d_in[4];
    const float* We1   = (const float*)d_in[5];
    const float* be1   = (const float*)d_in[6];
    const float* We2   = (const float*)d_in[7];
    const float* be2   = (const float*)d_in[8];
    const float* Winf  = (const float*)d_in[9];
    const float* binf  = (const float*)d_in[10];
    const float* Wx1   = (const float*)d_in[11];
    const float* bx1   = (const float*)d_in[12];
    const float* Wx2   = (const float*)d_in[13];
    const float* Wn1   = (const float*)d_in[14];
    const float* bn1   = (const float*)d_in[15];
    const float* Wn2   = (const float*)d_in[16];
    const float* bn2   = (const float*)d_in[17];

    // ---- workspace carve ----
    char* ws = (char*)d_ws;
    float* mi  = (float*)ws;                                   // [NN*HIDN] f32
    float* dxb = (float*)(ws + (size_t)NN * HIDN * 4);         // [NN*3] f32
    size_t off = (size_t)NN * HIDN * 4 + (size_t)NN * 3 * 4;   // 16B aligned
    uint32_t* pWe1 = (uint32_t*)(ws + off); off += (size_t)9 * 8 * 256 * 4;
    uint32_t* pWe2 = (uint32_t*)(ws + off); off += (size_t)4 * 8 * 256 * 4;
    uint32_t* pWx1 = (uint32_t*)(ws + off); off += (size_t)4 * 8 * 256 * 4;
    uint32_t* pWn1 = (uint32_t*)(ws + off); off += (size_t)8 * 8 * 256 * 4;
    uint32_t* pWn2 = (uint32_t*)(ws + off); off += (size_t)4 * 8 * 256 * 4;

    // zero accumulators (mi and dxb are contiguous)
    const int nzero = NN * HIDN + NN * 3;
    zero_kernel<<<(nzero + 255) / 256, 256, 0, stream>>>(mi, nzero);

    // pack weights fp32 -> bf16 WMMA-B fragments
    pack_w<<<(9 * 8 * 256 + 255) / 256, 256, 0, stream>>>(We1, pWe1, 280, HIDN, 9, 8);
    pack_w<<<(4 * 8 * 256 + 255) / 256, 256, 0, stream>>>(We2, pWe2, 128, HIDN, 4, 8);
    pack_w<<<(4 * 8 * 256 + 255) / 256, 256, 0, stream>>>(Wx1, pWx1, 128, HIDN, 4, 8);
    pack_w<<<(8 * 8 * 256 + 255) / 256, 256, 0, stream>>>(Wn1, pWn1, 256, HIDN, 8, 8);
    pack_w<<<(4 * 8 * 256 + 255) / 256, 256, 0, stream>>>(Wn2, pWn2, 128, HIDN, 4, 8);

    // edge phase
    edge_kernel<<<NE / ETILE, 128, 0, stream>>>(
        h, x, ei, eattr, be1, be2, Winf, binf, bx1, Wx2,
        pWe1, pWe2, pWx1, mi, dxb);

    // node phase
    float* h_out = (float*)d_out;
    float* x_out = h_out + (size_t)NN * HIDN;
    node_kernel<<<(NN + NTILE - 1) / NTILE, 128, 0, stream>>>(
        h, x, mask, bn1, bn2, pWn1, pWn2, mi, dxb, h_out, x_out);
}
